// GraphModule_12189117186589
// MI455X (gfx1250) — compile-verified, matrix operand order checked by reading, not run on hardware
//
#include <hip/hip_runtime.h>

// ---------------------------------------------------------------------------
// Types for CDNA5 WMMA (gfx1250, wave32)
// ---------------------------------------------------------------------------
typedef unsigned short bf16u;
typedef __attribute__((ext_vector_type(8)))  __bf16 v8bf;
typedef __attribute__((ext_vector_type(16))) __bf16 v16bf;
typedef __attribute__((ext_vector_type(8)))  float  v8f;

// Problem constants
#define CL  4
#define CB  16
#define CT  512
#define CD  512
#define CH  8
#define CDK 64
#define CDF 2048
#define CM  (CB*CT)          // 8192 rows

__device__ __forceinline__ bf16u f2bf(float f) {
    union { float f; unsigned u; } c; c.f = f;
    unsigned u = c.u;
    unsigned r = u + 0x7FFFu + ((u >> 16) & 1u);   // round-to-nearest-even
    return (bf16u)(r >> 16);
}

// Load one 16x32 bf16 fragment (A or B^T form), row-major, row stride ld.
// Per ISA 16-bit matrix layout: lane = m + 16*half, elems 0..7 = K[half*8..],
// elems 8..15 = K[16+half*8..].
__device__ __forceinline__ v16bf ldfrag(const bf16u* base, int ld, int r, int hf) {
    const bf16u* q = base + (long long)r * ld + hf * 8;
    v8bf lo = *(const v8bf*)(q);
    v8bf hi = *(const v8bf*)(q + 16);
    return __builtin_shufflevector(lo, hi, 0,1,2,3,4,5,6,7,8,9,10,11,12,13,14,15);
}

// ---------------------------------------------------------------------------
// Generic bf16 WMMA GEMM:  C[M,N] = A[M,K] * B[N,K]^T  (both K-contiguous)
// Block tile 64 x BN; wave tile 32x32 (2 A-frags + 2 B-frags -> 4 WMMAs,
// 1 fragment load per WMMA). BN=128 -> 8 waves, BN=64 -> 4 waves.
// Inner loop is manually software-pipelined: next k-tile's fragments are
// loaded before the current tile's WMMAs issue, so loadcnt waits overlap.
// ---------------------------------------------------------------------------
enum { EPI_F32 = 0, EPI_BD = 1, EPI_BF = 2, EPI_DUAL = 3, EPI_VT = 4, EPI_RES = 5 };

struct GemmP {
    const bf16u* A; const bf16u* B;
    int lda, ldb, ldc, K;
    int zdiv;                                   // z -> (zb = z/zdiv, zh = z%zdiv)
    long long sAb, sAh, sBb, sBh, sCb, sCh;     // batch offsets (elements)
    float* outF;
    bf16u* outBF; bf16u* outBF2;
    const float* bias; const float* bias2a; const float* bias2b;
    const float* residual;
    int relu;
};

template <int EPI, int BN>
__global__ __launch_bounds__(256) void gemm_k(GemmP p) {
    int z  = blockIdx.z;
    int zb = z / p.zdiv, zh = z % p.zdiv;
    const bf16u* A  = p.A + zb * p.sAb + zh * p.sAh;
    const bf16u* Bm = p.B + zb * p.sBb + zh * p.sBh;
    long long coff  = zb * p.sCb + zh * p.sCh;

    int tm = blockIdx.y * 64, tn = blockIdx.x * BN;
    int w = threadIdx.x >> 5, lane = threadIdx.x & 31;
    int wm = (w & 1) * 32, wn = (w >> 1) * 32;
    int r = lane & 15, hf = lane >> 4;

    v8f acc[2][2];
    #pragma unroll
    for (int i = 0; i < 8; ++i) {
        acc[0][0][i] = 0.f; acc[0][1][i] = 0.f;
        acc[1][0][i] = 0.f; acc[1][1][i] = 0.f;
    }

    const bf16u* Arow0 = A  + (long long)(tm + wm) * p.lda;
    const bf16u* Arow1 = Arow0 + (long long)16 * p.lda;
    const bf16u* Brow0 = Bm + (long long)(tn + wn) * p.ldb;
    const bf16u* Brow1 = Brow0 + (long long)16 * p.ldb;
    long long lo = (long long)r;   // lane row within fragment

    // prologue: fragments for k = 0
    v16bf a0 = ldfrag(Arow0, p.lda, r, hf);
    v16bf a1 = ldfrag(Arow1, p.lda, r, hf);
    v16bf b0 = ldfrag(Brow0, p.ldb, r, hf);
    v16bf b1 = ldfrag(Brow1, p.ldb, r, hf);

    for (int k = 32; k < p.K; k += 32) {
        // load next k-tile first (overlaps with WMMAs below)
        v16bf na0 = ldfrag(Arow0 + k, p.lda, r, hf);
        v16bf na1 = ldfrag(Arow1 + k, p.lda, r, hf);
        v16bf nb0 = ldfrag(Brow0 + k, p.ldb, r, hf);
        v16bf nb1 = ldfrag(Brow1 + k, p.ldb, r, hf);
        if (k + 96 < p.K) {   // prefetch ahead into near caches
            __builtin_prefetch(Arow0 + k + 96 + lo * p.lda, 0, 3);
            __builtin_prefetch(Brow0 + k + 96 + lo * p.ldb, 0, 3);
        }
        acc[0][0] = __builtin_amdgcn_wmma_f32_16x16x32_bf16(false, a0, false, b0,
                                                            (short)0, acc[0][0], false, false);
        acc[1][0] = __builtin_amdgcn_wmma_f32_16x16x32_bf16(false, a1, false, b0,
                                                            (short)0, acc[1][0], false, false);
        acc[0][1] = __builtin_amdgcn_wmma_f32_16x16x32_bf16(false, a0, false, b1,
                                                            (short)0, acc[0][1], false, false);
        acc[1][1] = __builtin_amdgcn_wmma_f32_16x16x32_bf16(false, a1, false, b1,
                                                            (short)0, acc[1][1], false, false);
        a0 = na0; a1 = na1; b0 = nb0; b1 = nb1;
    }
    // epilogue k-tile
    acc[0][0] = __builtin_amdgcn_wmma_f32_16x16x32_bf16(false, a0, false, b0,
                                                        (short)0, acc[0][0], false, false);
    acc[1][0] = __builtin_amdgcn_wmma_f32_16x16x32_bf16(false, a1, false, b0,
                                                        (short)0, acc[1][0], false, false);
    acc[0][1] = __builtin_amdgcn_wmma_f32_16x16x32_bf16(false, a0, false, b1,
                                                        (short)0, acc[0][1], false, false);
    acc[1][1] = __builtin_amdgcn_wmma_f32_16x16x32_bf16(false, a1, false, b1,
                                                        (short)0, acc[1][1], false, false);

    #pragma unroll
    for (int sn = 0; sn < 2; ++sn) {
        int col = tn + wn + sn * 16 + r;
        #pragma unroll
        for (int sm = 0; sm < 2; ++sm) {
            #pragma unroll
            for (int rr = 0; rr < 8; ++rr) {
                int row  = tm + wm + sm * 16 + rr + 8 * hf;  // C layout: VGPR rr -> M=rr / rr+8
                float val = acc[sm][sn][rr];
                if (EPI == EPI_F32) {
                    p.outF[coff + (long long)row * p.ldc + col] = val;
                } else if (EPI == EPI_BD) {
                    // rel_shift scatter-add: bd[t,rI] -> scores
                    int t = row, rI = col;
                    if (rI >= (CT - 1) - t) {
                        int s = rI - ((CT - 1) - t);
                        p.outF[coff + (long long)t * CT + s] += val;
                    } else if (t >= 1) {
                        int s = rI + t + 1;
                        p.outF[coff + (long long)(t - 1) * CT + s] += val;
                    }
                } else if (EPI == EPI_BF) {
                    float v2 = val + (p.bias ? p.bias[col] : 0.f);
                    if (p.relu) v2 = fmaxf(v2, 0.f);
                    p.outBF[coff + (long long)row * p.ldc + col] = f2bf(v2);
                } else if (EPI == EPI_DUAL) {
                    float base = val + p.bias[col];
                    long long idx = coff + (long long)row * p.ldc + col;
                    p.outBF [idx] = f2bf(base + p.bias2a[col]);
                    p.outBF2[idx] = f2bf(base + p.bias2b[col]);
                } else if (EPI == EPI_VT) {
                    // store V transposed: [B, H, DK, T]
                    int b = row >> 9, t = row & (CT - 1);
                    int h = col >> 6, dk = col & (CDK - 1);
                    long long idx = (((long long)b * CH + h) * CDK + dk) * CT + t;
                    p.outBF[idx] = f2bf(val + p.bias[col]);
                } else { // EPI_RES
                    long long idx = coff + (long long)row * p.ldc + col;
                    p.outF[idx] = p.residual[idx] + val + p.bias[col];
                }
            }
        }
    }
}

// ---------------------------------------------------------------------------
// LayerNorm over D=512 (one block of 128 threads per row); bf16 or f32 out.
// ---------------------------------------------------------------------------
__global__ __launch_bounds__(128) void ln_k(const float* __restrict__ x,
                                            const float* __restrict__ w,
                                            const float* __restrict__ b,
                                            bf16u* __restrict__ obf,
                                            float* __restrict__ of) {
    __shared__ float red[128];
    long long row = blockIdx.x;
    const float* src = x + row * CD;
    int tid = threadIdx.x;
    float l0 = src[tid], l1 = src[tid + 128], l2 = src[tid + 256], l3 = src[tid + 384];
    red[tid] = l0 + l1 + l2 + l3;
    __syncthreads();
    for (int st = 64; st > 0; st >>= 1) { if (tid < st) red[tid] += red[tid + st]; __syncthreads(); }
    float mu = red[0] * (1.f / CD);
    __syncthreads();
    float d0 = l0 - mu, d1 = l1 - mu, d2 = l2 - mu, d3 = l3 - mu;
    red[tid] = d0 * d0 + d1 * d1 + d2 * d2 + d3 * d3;
    __syncthreads();
    for (int st = 64; st > 0; st >>= 1) { if (tid < st) red[tid] += red[tid + st]; __syncthreads(); }
    float rs = rsqrtf(red[0] * (1.f / CD) + 1e-5f);
    float dd[4] = { d0, d1, d2, d3 };
    #pragma unroll
    for (int i = 0; i < 4; ++i) {
        int c = tid + i * 128;
        float o = dd[i] * rs * w[c] + b[c];
        if (obf) obf[row * CD + c] = f2bf(o);
        else     of [row * CD + c] = o;
    }
}

// ---------------------------------------------------------------------------
// Masked scaled softmax over s (row = b*H*T + h*T + t), f32 in -> bf16 out.
// ---------------------------------------------------------------------------
__global__ __launch_bounds__(128) void softmax_k(const float* __restrict__ scores,
                                                 bf16u* __restrict__ attn,
                                                 const unsigned char* __restrict__ mask,
                                                 float scale) {
    __shared__ float red[128];
    long long row = blockIdx.x;
    int b = (int)(row >> 12);                 // row / (H*T)
    const float* src = scores + row * CT;
    bf16u* dst = attn + row * CT;
    int tid = threadIdx.x;

    float v[4]; unsigned char mk[4];
    float mx = -3.40282347e38f;
    #pragma unroll
    for (int i = 0; i < 4; ++i) {
        int s = tid + i * 128;
        mk[i] = mask[b * CT + s];
        float x = src[s] * scale;
        v[i] = mk[i] ? x : -3.40282347e38f;
        mx = fmaxf(mx, v[i]);
    }
    red[tid] = mx; __syncthreads();
    for (int st = 64; st > 0; st >>= 1) { if (tid < st) red[tid] = fmaxf(red[tid], red[tid + st]); __syncthreads(); }
    float m = red[0];
    __syncthreads();
    float e[4]; float sum = 0.f;
    #pragma unroll
    for (int i = 0; i < 4; ++i) { e[i] = __expf(v[i] - m); sum += e[i]; }
    red[tid] = sum; __syncthreads();
    for (int st = 64; st > 0; st >>= 1) { if (tid < st) red[tid] += red[tid + st]; __syncthreads(); }
    float inv = 1.f / red[0];
    #pragma unroll
    for (int i = 0; i < 4; ++i) {
        float a = mk[i] ? e[i] * inv : 0.f;
        dst[tid + i * 128] = f2bf(a);
    }
}

// ---------------------------------------------------------------------------
// Elementwise helpers
// ---------------------------------------------------------------------------
__global__ void cvt_k(const float* __restrict__ in, bf16u* __restrict__ out, int n) {
    int i = blockIdx.x * 256 + threadIdx.x;
    if (i < n) out[i] = f2bf(in[i]);
}
__global__ void copy_k(const float* __restrict__ in, float* __restrict__ out, int n) {
    int i = blockIdx.x * 256 + threadIdx.x;
    if (i < n) out[i] = in[i];
}

// ---------------------------------------------------------------------------
template <int EPI, int BN>
static void run_gemm(const GemmP& p, int M, int N, int Z, hipStream_t s) {
    dim3 grid(N / BN, M / 64, Z);
    dim3 block(BN == 128 ? 256 : 128);
    gemm_k<EPI, BN><<<grid, block, 0, s>>>(p);
}

extern "C" void kernel_launch(void* const* d_in, const int* in_sizes, int n_in,
                              void* d_out, int out_size, void* d_ws, size_t ws_size,
                              hipStream_t stream) {
    (void)in_sizes; (void)n_in; (void)out_size; (void)ws_size;

    const unsigned char* mask = (const unsigned char*)d_in[0];
    const float* xs       = (const float*)d_in[1];
    const float* pos_emb  = (const float*)d_in[2];
    const float* ln_mha_w = (const float*)d_in[3];
    const float* ln_mha_b = (const float*)d_in[4];
    const float* wq       = (const float*)d_in[5];
    const float* bq       = (const float*)d_in[6];
    const float* wk       = (const float*)d_in[7];
    const float* bk       = (const float*)d_in[8];
    const float* wv       = (const float*)d_in[9];
    const float* bv       = (const float*)d_in[10];
    const float* wp       = (const float*)d_in[11];
    const float* pbu      = (const float*)d_in[12];
    const float* pbv      = (const float*)d_in[13];
    const float* wo       = (const float*)d_in[14];
    const float* bo       = (const float*)d_in[15];
    const float* ln_ff_w  = (const float*)d_in[16];
    const float* ln_ff_b  = (const float*)d_in[17];
    const float* w1       = (const float*)d_in[18];
    const float* b1       = (const float*)d_in[19];
    const float* w2       = (const float*)d_in[20];
    const float* b2       = (const float*)d_in[21];
    const float* anw      = (const float*)d_in[22];
    const float* anb      = (const float*)d_in[23];

    // ---- workspace carve ----
    char* ws = (char*)d_ws;
    size_t off = 0;
    auto alloc = [&](size_t bytes) -> void* {
        void* p = ws + off; off += (bytes + 255) & ~(size_t)255; return p;
    };
    const int DD = CD * CD;
    bf16u* wq_bf  = (bf16u*)alloc((size_t)CL * DD * 2);
    bf16u* wk_bf  = (bf16u*)alloc((size_t)CL * DD * 2);
    bf16u* wv_bf  = (bf16u*)alloc((size_t)CL * DD * 2);
    bf16u* wp_bf  = (bf16u*)alloc((size_t)CL * DD * 2);
    bf16u* wo_bf  = (bf16u*)alloc((size_t)CL * DD * 2);
    bf16u* w1_bf  = (bf16u*)alloc((size_t)CL * CDF * CD * 2);
    bf16u* w2_bf  = (bf16u*)alloc((size_t)CL * CD * CDF * 2);
    bf16u* pos_bf = (bf16u*)alloc((size_t)CT * CD * 2);
    float* x      = (float*)alloc((size_t)CM * CD * 4);
    bf16u* xn_bf  = (bf16u*)alloc((size_t)CM * CD * 2);
    bf16u* qu_bf  = (bf16u*)alloc((size_t)CM * CD * 2);
    bf16u* qv_bf  = (bf16u*)alloc((size_t)CM * CD * 2);
    bf16u* k_bf   = (bf16u*)alloc((size_t)CM * CD * 2);
    bf16u* vT_bf  = (bf16u*)alloc((size_t)CM * CD * 2);
    bf16u* p_bf   = (bf16u*)alloc((size_t)CT * CD * 2);
    bf16u* ctx_bf = (bf16u*)alloc((size_t)CM * CD * 2);
    bf16u* ffh_bf = (bf16u*)alloc((size_t)CM * CDF * 2);
    float* scores = (float*)alloc((size_t)CB * CH * CT * CT * 4);
    bf16u* attnbf = (bf16u*)alloc((size_t)CB * CH * CT * CT * 2);

    auto cvt = [&](const float* src, bf16u* dst, int n) {
        cvt_k<<<(n + 255) / 256, 256, 0, stream>>>(src, dst, n);
    };
    // ---- one-time conversions ----
    cvt(wq, wq_bf, CL * DD);     cvt(wk, wk_bf, CL * DD);
    cvt(wv, wv_bf, CL * DD);     cvt(wp, wp_bf, CL * DD);
    cvt(wo, wo_bf, CL * DD);
    cvt(w1, w1_bf, CL * CDF * CD);
    cvt(w2, w2_bf, CL * CD * CDF);
    cvt(pos_emb, pos_bf, CT * CD);
    copy_k<<<(CM * CD + 255) / 256, 256, 0, stream>>>(xs, x, CM * CD);

    const long long sTD = (long long)CT * CD;     // 262144
    const long long sTT = (long long)CT * CT;     // 262144

    for (int i = 0; i < CL; ++i) {
        // ---- MHA layernorm ----
        ln_k<<<CM, 128, 0, stream>>>(x, ln_mha_w + i * CD, ln_mha_b + i * CD, xn_bf, nullptr);

        // Q projection: dual output q+u, q+v
        GemmP g{};
        g.lda = CD; g.ldb = CD; g.ldc = CD; g.K = CD; g.zdiv = 1;
        g.A = xn_bf; g.B = wq_bf + (size_t)i * DD;
        g.bias = bq + i * CD; g.bias2a = pbu + i * CD; g.bias2b = pbv + i * CD;
        g.outBF = qu_bf; g.outBF2 = qv_bf;
        run_gemm<EPI_DUAL, 128>(g, CM, CD, 1, stream);

        // K projection
        GemmP gk{}; gk.lda = CD; gk.ldb = CD; gk.ldc = CD; gk.K = CD; gk.zdiv = 1;
        gk.A = xn_bf; gk.B = wk_bf + (size_t)i * DD; gk.bias = bk + i * CD; gk.outBF = k_bf;
        run_gemm<EPI_BF, 128>(gk, CM, CD, 1, stream);

        // V projection (transposed store -> [B,H,DK,T])
        GemmP gv{}; gv.lda = CD; gv.ldb = CD; gv.ldc = CD; gv.K = CD; gv.zdiv = 1;
        gv.A = xn_bf; gv.B = wv_bf + (size_t)i * DD; gv.bias = bv + i * CD; gv.outBF = vT_bf;
        run_gemm<EPI_VT, 128>(gv, CM, CD, 1, stream);

        // P projection (no bias)
        GemmP gp{}; gp.lda = CD; gp.ldb = CD; gp.ldc = CD; gp.K = CD; gp.zdiv = 1;
        gp.A = pos_bf; gp.B = wp_bf + (size_t)i * DD; gp.outBF = p_bf;
        run_gemm<EPI_BF, 128>(gp, CT, CD, 1, stream);

        // matrix_ac: scores[b,h,t,s] = sum_d qu[b,t,h,d] k[b,s,h,d]
        GemmP ga{}; ga.lda = CD; ga.ldb = CD; ga.ldc = CT; ga.K = CDK; ga.zdiv = CH;
        ga.A = qu_bf; ga.sAb = sTD; ga.sAh = CDK;
        ga.B = k_bf;  ga.sBb = sTD; ga.sBh = CDK;
        ga.outF = scores; ga.sCb = (long long)CH * sTT; ga.sCh = sTT;
        run_gemm<EPI_F32, 128>(ga, CT, CT, CB * CH, stream);

        // matrix_bd + rel_shift scatter-add into scores
        GemmP gb{}; gb.lda = CD; gb.ldb = CD; gb.ldc = CT; gb.K = CDK; gb.zdiv = CH;
        gb.A = qv_bf; gb.sAb = sTD; gb.sAh = CDK;
        gb.B = p_bf;  gb.sBb = 0;   gb.sBh = CDK;
        gb.outF = scores; gb.sCb = (long long)CH * sTT; gb.sCh = sTT;
        run_gemm<EPI_BD, 128>(gb, CT, CT, CB * CH, stream);

        // softmax (scale + mask) -> bf16 attn
        softmax_k<<<CB * CH * CT, 128, 0, stream>>>(scores, attnbf, mask, 0.125f);

        // ctx[b,t,h,dk] = sum_s attn[b,h,t,s] vT[b,h,dk,s]
        GemmP gc{}; gc.lda = CT; gc.ldb = CT; gc.ldc = CD; gc.K = CT; gc.zdiv = CH;
        gc.A = attnbf; gc.sAb = (long long)CH * sTT; gc.sAh = sTT;
        gc.B = vT_bf;  gc.sBb = (long long)CH * CDK * CT; gc.sBh = (long long)CDK * CT;
        gc.outBF = ctx_bf; gc.sCb = sTD; gc.sCh = CDK;
        run_gemm<EPI_BF, 64>(gc, CT, CDK, CB * CH, stream);

        // x = x + ctx @ wo^T + bo
        GemmP go{}; go.lda = CD; go.ldb = CD; go.ldc = CD; go.K = CD; go.zdiv = 1;
        go.A = ctx_bf; go.B = wo_bf + (size_t)i * DD; go.bias = bo + i * CD;
        go.residual = x; go.outF = x;
        run_gemm<EPI_RES, 128>(go, CM, CD, 1, stream);

        // ---- FFN ----
        ln_k<<<CM, 128, 0, stream>>>(x, ln_ff_w + i * CD, ln_ff_b + i * CD, xn_bf, nullptr);

        GemmP g1{}; g1.lda = CD; g1.ldb = CD; g1.ldc = CDF; g1.K = CD; g1.zdiv = 1;
        g1.A = xn_bf; g1.B = w1_bf + (size_t)i * CDF * CD; g1.bias = b1 + i * CDF;
        g1.outBF = ffh_bf; g1.relu = 1;
        run_gemm<EPI_BF, 128>(g1, CM, CDF, 1, stream);

        GemmP g2{}; g2.lda = CDF; g2.ldb = CDF; g2.ldc = CD; g2.K = CDF; g2.zdiv = 1;
        g2.A = ffh_bf; g2.B = w2_bf + (size_t)i * CD * CDF; g2.bias = b2 + i * CD;
        g2.residual = x; g2.outF = x;
        run_gemm<EPI_RES, 128>(g2, CM, CD, 1, stream);
    }

    // final layernorm -> d_out (fp32)
    ln_k<<<CM, 128, 0, stream>>>(x, anw, anb, nullptr, (float*)d_out);
}
